// MultiLatentSpaceSimilarity_34033320854189
// MI455X (gfx1250) — matrix operand order repaired and checked by reading, not compiled, not run on hardware
//
#include <hip/hip_runtime.h>

typedef __attribute__((ext_vector_type(16))) _Float16 v16h;
typedef __attribute__((ext_vector_type(8)))  float    v8f;
typedef unsigned __attribute__((ext_vector_type(4))) uv4;
typedef int      __attribute__((ext_vector_type(8))) iv8;
typedef int      __attribute__((ext_vector_type(4))) iv4;

#define B_     16
#define C_     256
#define S_     1024
#define PTOT   24
#define L_     64
#define HEADS_ 8
#define DH_    64
#define INNER_ 512
#define SPLITS_ 8
#define SCALE_ 0.125f

#define SS_PITCH 1032              // halfs; 516 dwords -> conflict-free row stride
#define SQ_PITCH 72                // halfs; 36 dwords
#define SQ_LDS_OFF (64u * SS_PITCH * 2u)   // byte offset of sQ in dynamic LDS

// ---------------- WMMA fragment gathers (wave32, documented CDNA5 layouts) ---------

// A matrix 16x32 f16. Lane L: M = mbase + (L&15). Pair e (VGPR e) holds K pair:
//  half = L>>4;  e<4 : K = half*8 + 2e ;  e>=4 : K = 16 + half*8 + 2(e-4)
__device__ __forceinline__ v16h frag_a(const _Float16* base, int pitch, int mbase, int k0) {
    int lane = threadIdx.x & 31;
    int mm = mbase + (lane & 15);
    int hf = lane >> 4;
    const _Float16* rp = base + (long)mm * pitch + k0;
    union { v16h v; unsigned u[8]; } r;
#pragma unroll
    for (int e = 0; e < 8; ++e) {
        int K = (e < 4) ? (hf * 8 + 2 * e) : (16 + hf * 8 + 2 * (e - 4));
        r.u[e] = *(const unsigned*)(rp + K);
    }
    return r.v;
}

// B matrix 32x16 f16. Lane L: N = nbase + (L&15); K pair e: half = L>>4; e<4 : K = half*8 + 2e ; e>=4 : K = 16 + half*8 + 2(e-4)
__device__ __forceinline__ v16h frag_b(const _Float16* base, int pitch, int nbase, int k0) {
    int lane = threadIdx.x & 31;
    int nn = nbase + (lane & 15);
    int hf = lane >> 4;
    const _Float16* rp = base + (long)nn * pitch + k0;
    union { v16h v; unsigned u[8]; } r;
#pragma unroll
    for (int e = 0; e < 8; ++e) r.u[e] = *(const unsigned*)(rp + ((e < 4) ? (hf * 8 + 2 * e) : (16 + hf * 8 + 2 * (e - 4))));
    return r.v;
}

// ---------------- init: zero accumulators ----------------
__global__ __launch_bounds__(256) void k_init(float* heat, float* dist) {
    int n = B_ * PTOT * S_;
    for (int i = blockIdx.x * 256 + threadIdx.x; i < n; i += gridDim.x * 256) heat[i] = 0.f;
    for (int i = blockIdx.x * 256 + threadIdx.x; i < B_ * PTOT; i += gridDim.x * 256) dist[i] = 0.f;
}

// ---------------- prototype Q projection (fp32 exact; tiny: 0.4 GFLOP) ------------
__global__ __launch_bounds__(256) void k_protoq(const float* __restrict__ prot,
                                                const float* __restrict__ Wq,
                                                float* __restrict__ q32,
                                                _Float16* __restrict__ q16) {
    int pg = blockIdx.x;
    int n0 = blockIdx.y * 64;
    int wi = pg >> 3;
    __shared__ float sP[32][64];
    __shared__ float sW[32][64];
    int t = threadIdx.x;
    int l = t >> 2;
    int nb = t & 3;
    float acc[16];
#pragma unroll
    for (int j = 0; j < 16; ++j) acc[j] = 0.f;
    for (int c0 = 0; c0 < C_; c0 += 32) {
        __syncthreads();
#pragma unroll
        for (int k = 0; k < 8; ++k) {
            int idx = t + k * 256;
            int c = idx >> 6, e = idx & 63;
            sP[c][e] = prot[((long)pg * C_ + c0 + c) * L_ + e];
            sW[c][e] = Wq[((long)wi * C_ + c0 + c) * INNER_ + n0 + e];
        }
        __syncthreads();
        for (int c = 0; c < 32; ++c) {
            float pl = sP[c][l];
#pragma unroll
            for (int j = 0; j < 16; ++j) acc[j] += pl * sW[c][nb + 4 * j];
        }
    }
#pragma unroll
    for (int j = 0; j < 16; ++j) {
        long o = ((long)pg * L_ + l) * INNER_ + n0 + nb + 4 * j;
        q32[o] = acc[j];
        q16[o] = (_Float16)acc[j];
    }
}

// ---------------- K/V projection GEMM (f16 WMMA, fp32 acc) ------------------------
__global__ __launch_bounds__(256) void k_kv(const float* __restrict__ x,
                                            const float* __restrict__ Wk,
                                            const float* __restrict__ Wv,
                                            _Float16* __restrict__ k16,
                                            _Float16* __restrict__ v16T, int wi) {
    int b = blockIdx.x;
    int s0 = blockIdx.y * 64;
    int n0 = blockIdx.z * 64;
    __shared__ _Float16 sA[64][36];   // xT tile [s][c]
    __shared__ _Float16 sBk[64][36];  // Wk tile [n][c]
    __shared__ _Float16 sBv[64][36];  // Wv tile [n][c]
    int t = threadIdx.x, lane = t & 31, w = t >> 5;
    int mt = (w & 3) * 16;
    int ntA = (w >> 2) * 16;
    int ntB = ntA + 32;
    v8f aK0 = {0,0,0,0,0,0,0,0}, aK1 = {0,0,0,0,0,0,0,0};
    v8f aV0 = {0,0,0,0,0,0,0,0}, aV1 = {0,0,0,0,0,0,0,0};
    const float* xb = x + (long)b * C_ * S_;
    const float* wkp = Wk + (long)wi * C_ * INNER_;
    const float* wvp = Wv + (long)wi * C_ * INNER_;

    for (int c0 = 0; c0 < C_; c0 += 32) {
        __syncthreads();
#pragma unroll
        for (int k = 0; k < 4; ++k) {
            int idx = t + k * 256;
            int e  = idx & 63;           // s-local / n-local (coalesced)
            int cp = (idx >> 6) * 2;     // even c pair
            union { unsigned u; _Float16 h[2]; } pa, pb, pc;
            pa.h[0] = (_Float16)xb[(long)(c0 + cp) * S_ + s0 + e];
            pa.h[1] = (_Float16)xb[(long)(c0 + cp + 1) * S_ + s0 + e];
            *(unsigned*)&sA[e][cp] = pa.u;
            pb.h[0] = (_Float16)wkp[(long)(c0 + cp) * INNER_ + n0 + e];
            pb.h[1] = (_Float16)wkp[(long)(c0 + cp + 1) * INNER_ + n0 + e];
            *(unsigned*)&sBk[e][cp] = pb.u;
            pc.h[0] = (_Float16)wvp[(long)(c0 + cp) * INNER_ + n0 + e];
            pc.h[1] = (_Float16)wvp[(long)(c0 + cp + 1) * INNER_ + n0 + e];
            *(unsigned*)&sBv[e][cp] = pc.u;
        }
        __syncthreads();
        v16h af  = frag_a(&sA[0][0], 36, mt, 0);
        v16h bk0 = frag_b(&sBk[0][0], 36, ntA, 0);
        v16h bk1 = frag_b(&sBk[0][0], 36, ntB, 0);
        v16h bv0 = frag_b(&sBv[0][0], 36, ntA, 0);
        v16h bv1 = frag_b(&sBv[0][0], 36, ntB, 0);
        aK0 = __builtin_amdgcn_wmma_f32_16x16x32_f16(false, af, false, bk0, (short)0, aK0, false, false);
        aK1 = __builtin_amdgcn_wmma_f32_16x16x32_f16(false, af, false, bk1, (short)0, aK1, false, false);
        aV0 = __builtin_amdgcn_wmma_f32_16x16x32_f16(false, af, false, bv0, (short)0, aV0, false, false);
        aV1 = __builtin_amdgcn_wmma_f32_16x16x32_f16(false, af, false, bv1, (short)0, aV1, false, false);
    }
    int hf = lane >> 4, ncol = lane & 15;
#pragma unroll
    for (int r = 0; r < 8; ++r) {
        int s = s0 + mt + hf * 8 + r;
        int n = n0 + ntA + ncol;
        int h = n >> 6, d = n & 63;
        k16[(((long)b * HEADS_ + h) * S_ + s) * DH_ + d] = (_Float16)aK0[r];
        v16T[(((long)b * HEADS_ + h) * DH_ + d) * S_ + s] = (_Float16)aV0[r];
        n = n0 + ntB + ncol; h = n >> 6; d = n & 63;
        k16[(((long)b * HEADS_ + h) * S_ + s) * DH_ + d] = (_Float16)aK1[r];
        v16T[(((long)b * HEADS_ + h) * DH_ + d) * S_ + s] = (_Float16)aV1[r];
    }
}

// ---------------- fused attention: QK^T -> softmax -> PV, heat & dist -------------
__global__ __launch_bounds__(256) void k_attn(const _Float16* __restrict__ q16,
                                              const float* __restrict__ q32,
                                              const _Float16* __restrict__ k16,
                                              const _Float16* __restrict__ v16T,
                                              float* __restrict__ heat,
                                              float* __restrict__ dist, int wi) {
    int b = blockIdx.x, h = blockIdx.y, p = blockIdx.z;
    int pg = wi * SPLITS_ + p;
    extern __shared__ char smem[];
    _Float16* sS = (_Float16*)smem;               // [64][1032]
    _Float16* sQ = sS + 64 * SS_PITCH;            // [64][72]
    float* red    = (float*)(sQ + 64 * SQ_PITCH); // [256]
    float* rowmax = red + 256;                    // [64]
    float* rowinv = red + 320;                    // [64]
    int t = threadIdx.x, lane = t & 31, w = t >> 5;
    const _Float16* kbase = k16 + ((long)b * HEADS_ + h) * S_ * DH_;
    const _Float16* vbase = v16T + ((long)b * HEADS_ + h) * DH_ * S_;
    const _Float16* qsrc = q16 + (long)pg * L_ * INNER_ + h * DH_;

#if __has_builtin(__builtin_amdgcn_tensor_load_to_lds) && __has_builtin(__builtin_amdgcn_s_wait_tensorcnt)
    // TDM: DMA the 64x128B Q tile into LDS; pad 16B per 128B row -> pitch 144B (= SQ_PITCH halfs)
    if (w == 0) {
        unsigned long long ga = (unsigned long long)(const void*)qsrc;
        uv4 g0;
        g0.x = 1u;                                   // count=1, user descriptor, no gather
        g0.y = SQ_LDS_OFF;                           // lds_addr (dynamic LDS base is offset 0)
        g0.z = (unsigned)(ga & 0xffffffffull);       // global_addr[31:0]
        g0.w = (unsigned)((ga >> 32) & 0x1ffffffull) | (2u << 30);  // addr[56:32] | type=2
        iv8 g1;
        g1[0] = (int)((2u << 16)        // data_size = 4B
                    | (1u << 20)        // pad_enable
                    | (4u << 22)        // pad_interval: 128B
                    | (3u << 25));      // pad_amount: 4 dwords = 16B
        g1[1] = (int)(32u << 16);       // tensor_dim0 = 32 (x 4B = 128B row)
        g1[2] = (int)(64u << 16);       // tensor_dim1 = 64 rows
        g1[3] = (int)(32u << 16);       // tile_dim0 = 32
        g1[4] = 64;                     // tile_dim1 = 64
        g1[5] = 256;                    // tensor_dim0_stride = 256 (x 4B = 1024B = INNER_ halfs)
        g1[6] = 0;
        g1[7] = 0;
        iv4 g2 = {0, 0, 0, 0}, g3 = {0, 0, 0, 0};
        iv8 g4 = {0, 0, 0, 0, 0, 0, 0, 0};
        __builtin_amdgcn_tensor_load_to_lds(g0, g1, g2, g3, g4, 0);
        __builtin_amdgcn_s_wait_tensorcnt(0);
    }
#else
#pragma unroll
    for (int k = 0; k < 8; ++k) {
        int idx = t + k * 256;
        int l = idx >> 5;
        int dp = idx & 31;
        const unsigned* src = (const unsigned*)(qsrc + (long)l * INNER_) + dp;
        *(unsigned*)&sQ[l * SQ_PITCH + dp * 2] = *src;
    }
#endif
    __syncthreads();

    // Phase 1: scores = (Q K^T) * SCALE -> sS f16
    int m = (w & 3) * 16;
    int shalf = (w >> 2) * 512;
    for (int nt = 0; nt < 32; ++nt) {
        int stile = shalf + nt * 16;
        __builtin_prefetch((const void*)(kbase + (long)((stile + 16) & 1023) * DH_ + (long)lane * DH_), 0, 0);
        v8f acc = {0,0,0,0,0,0,0,0};
#pragma unroll
        for (int kc = 0; kc < 2; ++kc) {
            v16h a  = frag_a(sQ, SQ_PITCH, m, kc * 32);
            v16h bf = frag_b(kbase, DH_, stile, kc * 32);   // K[s][d]: B[k=d][n=s]
            acc = __builtin_amdgcn_wmma_f32_16x16x32_f16(false, a, false, bf, (short)0, acc, false, false);
        }
        int hf = lane >> 4, col = stile + (lane & 15);
#pragma unroll
        for (int r = 0; r < 8; ++r)
            sS[(m + hf * 8 + r) * SS_PITCH + col] = (_Float16)(acc[r] * SCALE_);
    }
    __syncthreads();

    // Phase 2: softmax stats (fp32), packed b32 row scans
    {
        int r = t >> 2, seg = (t & 3) * 256;
        const unsigned* rp = (const unsigned*)(sS + r * SS_PITCH + seg);
        float mx = -3.0e38f;
        for (int c2 = 0; c2 < 128; ++c2) {
            union { unsigned u; _Float16 hh[2]; } pk; pk.u = rp[c2];
            mx = fmaxf(mx, fmaxf((float)pk.hh[0], (float)pk.hh[1]));
        }
        red[t] = mx;
    }
    __syncthreads();
    if (t < 64) rowmax[t] = fmaxf(fmaxf(red[4 * t], red[4 * t + 1]), fmaxf(red[4 * t + 2], red[4 * t + 3]));
    __syncthreads();
    {
        int r = t >> 2, seg = (t & 3) * 256;
        const unsigned* rp = (const unsigned*)(sS + r * SS_PITCH + seg);
        float mx = rowmax[r], sum = 0.f;
        for (int c2 = 0; c2 < 128; ++c2) {
            union { unsigned u; _Float16 hh[2]; } pk; pk.u = rp[c2];
            sum += __expf((float)pk.hh[0] - mx) + __expf((float)pk.hh[1] - mx);
        }
        red[t] = sum;
    }
    __syncthreads();
    if (t < 64) rowinv[t] = 1.0f / (red[4 * t] + red[4 * t + 1] + red[4 * t + 2] + red[4 * t + 3]);
    __syncthreads();
    // normalize in-place + heat column sums
#pragma unroll
    for (int j = 0; j < 4; ++j) {
        int c = t + j * 256;
        float hsum = 0.f;
        for (int r = 0; r < 64; ++r) {
            float pv = __expf((float)sS[r * SS_PITCH + c] - rowmax[r]) * rowinv[r];
            sS[r * SS_PITCH + c] = (_Float16)pv;
            hsum += pv;
        }
        atomicAdd(&heat[((long)b * PTOT + pg) * S_ + c], hsum * (1.0f / 512.0f));
    }
    __syncthreads();

    // Phase 3: out = P @ V  (A from sS, B from v16T[d][s]); then dist vs q32
    int nd0 = (w >> 2) * 32;
    v8f o0 = {0,0,0,0,0,0,0,0}, o1 = {0,0,0,0,0,0,0,0};
    for (int sc = 0; sc < 32; ++sc) {
        int k0 = sc * 32;
        v16h a  = frag_a(sS, SS_PITCH, m, k0);
        v16h b0 = frag_b(vbase, S_, nd0, k0);
        v16h b1 = frag_b(vbase, S_, nd0 + 16, k0);
        o0 = __builtin_amdgcn_wmma_f32_16x16x32_f16(false, a, false, b0, (short)0, o0, false, false);
        o1 = __builtin_amdgcn_wmma_f32_16x16x32_f16(false, a, false, b1, (short)0, o1, false, false);
    }
    float dsum = 0.f;
    int hf = lane >> 4, ncol = lane & 15;
#pragma unroll
    for (int r = 0; r < 8; ++r) {
        int l = m + hf * 8 + r;
        long qo = ((long)pg * L_ + l) * INNER_ + h * DH_;
        float d0 = q32[qo + nd0 + ncol] - o0[r];
        float d1 = q32[qo + nd0 + 16 + ncol] - o1[r];
        dsum += d0 * d0 + d1 * d1;
    }
#pragma unroll
    for (int off = 16; off > 0; off >>= 1) dsum += __shfl_down(dsum, off, 32);
    if (lane == 0) atomicAdd(&dist[b * PTOT + pg], dsum);
}

// ---------------- finalize: argmax(heat) + scaled dist -> d_out -------------------
__global__ __launch_bounds__(256) void k_final(const float* __restrict__ heat,
                                               const float* __restrict__ dist,
                                               float* __restrict__ outF,
                                               int* __restrict__ outI) {
    int g = blockIdx.x;                    // b*24 + pg
    const float* hrow = heat + (long)g * S_;
    int t = threadIdx.x;
    float bv = -1.0f; int bi = 0;
#pragma unroll
    for (int j = 0; j < 4; ++j) {
        int c = t + j * 256;
        float v = hrow[c];
        if (v > bv) { bv = v; bi = c; }
    }
    __shared__ float sv[256];
    __shared__ int   si[256];
    sv[t] = bv; si[t] = bi;
    __syncthreads();
    for (int s = 128; s > 0; s >>= 1) {
        if (t < s) {
            float ov = sv[t + s]; int oi = si[t + s];
            if (ov > sv[t] || (ov == sv[t] && oi < si[t])) { sv[t] = ov; si[t] = oi; }
        }
        __syncthreads();
    }
    if (t == 0) {
        outF[g] = dist[g] * (1.0f / 32768.0f);  // mean over L*INNER
        outI[g] = si[0];
    }
}

// ---------------- host launcher ---------------------------------------------------
extern "C" void kernel_launch(void* const* d_in, const int* in_sizes, int n_in,
                              void* d_out, int out_size, void* d_ws, size_t ws_size,
                              hipStream_t stream) {
    const float* x    = (const float*)d_in[0];
    const float* prot = (const float*)d_in[1];
    const float* Wq   = (const float*)d_in[2];
    const float* Wk   = (const float*)d_in[3];
    const float* Wv   = (const float*)d_in[4];

    char* ws = (char*)d_ws;
    size_t off = 0;
    float*    q32  = (float*)(ws + off);    off += (size_t)PTOT * L_ * INNER_ * 4;
    _Float16* q16  = (_Float16*)(ws + off); off += (size_t)PTOT * L_ * INNER_ * 2;
    _Float16* k16  = (_Float16*)(ws + off); off += (size_t)B_ * HEADS_ * S_ * DH_ * 2;
    _Float16* v16T = (_Float16*)(ws + off); off += (size_t)B_ * HEADS_ * S_ * DH_ * 2;
    float*    heat = (float*)(ws + off);    off += (size_t)B_ * PTOT * S_ * 4;
    float*    dist = (float*)(ws + off);    off += (size_t)B_ * PTOT * 4;

    k_init<<<512, 256, 0, stream>>>(heat, dist);
    k_protoq<<<dim3(PTOT, INNER_ / 64), 256, 0, stream>>>(prot, Wq, q32, q16);

    size_t attn_lds = (size_t)64 * SS_PITCH * 2 + (size_t)64 * SQ_PITCH * 2 + (256 + 64 + 64) * 4;
    for (int i = 0; i < 3; ++i) {
        k_kv<<<dim3(B_, S_ / 64, INNER_ / 64), 256, 0, stream>>>(x, Wk, Wv, k16, v16T, i);
        k_attn<<<dim3(B_, HEADS_, SPLITS_), 256, attn_lds, stream>>>(q16, q32, k16, v16T, heat, dist, i);
    }
    k_final<<<B_ * PTOT, 256, 0, stream>>>(heat, dist, (float*)d_out, ((int*)d_out) + B_ * PTOT);
}
